// CausalSelfAttention_30786325578101
// MI455X (gfx1250) — matrix operand order repaired and checked by reading, not compiled, not run on hardware
//
#include <hip/hip_runtime.h>
#include <hip/hip_bf16.h>

typedef __attribute__((ext_vector_type(16))) __bf16 v16bf;
typedef __attribute__((ext_vector_type(8)))  __bf16 v8bf;
typedef __attribute__((ext_vector_type(8)))  float  v8f;

#define ZERO8F {0.f,0.f,0.f,0.f,0.f,0.f,0.f,0.f}

// ---------------------------------------------------------------------------
// WMMA helpers (gfx1250: D = A(16x32 bf16) * B(32x16 bf16) + C(16x16 f32))
// ---------------------------------------------------------------------------
__device__ __forceinline__ v8f wmma_bf16(v16bf a, v16bf b, v8f c) {
  return __builtin_amdgcn_wmma_f32_16x16x32_bf16(
      /*neg_a=*/false, a, /*neg_b=*/false, b,
      /*c_mod=*/(short)0, c, /*reuse_a=*/false, /*reuse_b=*/false);
}

// A fragment: rows m0..m0+15, K-tile of 32. Row-major source, ld elems/row.
// Lane L: row = L%16, holds K in [8h,8h+8) and [16+8h,16+8h+8), h=L/16.
__device__ __forceinline__ v16bf load_a16x32(const __bf16* base, int ld, int lane) {
  const int m = lane & 15, h = lane >> 4;
  const __bf16* p = base + m * ld + 8 * h;
  v8bf lo = *(const v8bf*)(p);
  v8bf hi = *(const v8bf*)(p + 16);
  v16bf a;
#pragma unroll
  for (int e = 0; e < 8; ++e) { a[e] = lo[e]; a[8 + e] = hi[e]; }
  return a;
}

// Guarded A fragment (rows >= mmax or K >= kmax -> 0). base = matrix origin.
__device__ __forceinline__ v16bf load_a_guard(const __bf16* base, int ld, int lane,
                                              int row0, int mmax, int k0, int kmax) {
  const int m = row0 + (lane & 15), h = lane >> 4;
  v16bf a;
#pragma unroll
  for (int e = 0; e < 16; ++e) {
    int kk = (e < 8 ? e : e + 8) + 8 * h;
    int kg = k0 + kk;
    a[e] = (m < mmax && kg < kmax) ? base[m * ld + kg] : (__bf16)0.f;
  }
  return a;
}

// B fragment: K-tile of 32, cols n0..n0+15. Column-major source (K contiguous
// within a column), ld = column stride. Lane L: col = L%16, K in [8h,8h+8) and [16+8h,16+8h+8).
__device__ __forceinline__ v16bf load_b32x16(const __bf16* colbase, int ld, int lane) {
  const int n = lane & 15, h = lane >> 4;
  const __bf16* p = colbase + n * ld + 8 * h;
  v8bf lo = *(const v8bf*)(p);
  v8bf hi = *(const v8bf*)(p + 16);
  v16bf b;
#pragma unroll
  for (int e = 0; e < 8; ++e) { b[e] = lo[e]; b[8 + e] = hi[e]; }
  return b;
}

__device__ __forceinline__ v16bf load_b_guard(const __bf16* colbase, int ld, int lane,
                                              int k0, int kmax) {
  const int n = lane & 15, h = lane >> 4;
  v16bf b;
#pragma unroll
  for (int e = 0; e < 16; ++e) {
    int kg = k0 + (e < 8 ? e : e + 8) + 8 * h;
    b[e] = (kg < kmax) ? colbase[n * ld + kg] : (__bf16)0.f;
  }
  return b;
}

// ---------------------------------------------------------------------------
// f32 -> bf16 conversion
// ---------------------------------------------------------------------------
__global__ void cvt_f32_bf16(const float* __restrict__ src, __bf16* __restrict__ dst, int n) {
  int i = blockIdx.x * blockDim.x + threadIdx.x;
  if (i < n) dst[i] = (__bf16)src[i];
}

// ---------------------------------------------------------------------------
// Projection GEMM: out[m,n] = sum_k X[m,k] * W[n,k]
//   X: (8000 x 1024) bf16 row-major, W: (1024 x 1024) bf16 row-major
//   transposed==0: dst[(b*8+h)*2000 + t][d]  (q layout, d contiguous)
//   transposed==1: dst[(b*8+h)*128 + d][t]   (t contiguous, for chunked GEMM)
// One wave per 32x64 tile (2 A frags x 4 B frags -> 8 WMMA / k-step).
// Tiles: 250 x 16 = 4000 waves.
// ---------------------------------------------------------------------------
__global__ __launch_bounds__(256) void proj_gemm(const __bf16* __restrict__ X,
                                                 const __bf16* __restrict__ W,
                                                 __bf16* __restrict__ dst,
                                                 int transposed) {
  const int lane = threadIdx.x & 31;
  const int wave = threadIdx.x >> 5;
  const int flat = blockIdx.x * (blockDim.x >> 5) + wave;
  if (flat >= 250 * 16) return;
  const int mt = flat >> 4, nt = flat & 15;
  const int m0 = mt * 32, n0 = nt * 64;

  const __bf16* ap = X + (size_t)m0 * 1024;
  const __bf16* bp = W + (size_t)n0 * 1024;  // row n of W == column n of B-op

  v8f c[2][4];
#pragma unroll
  for (int mi = 0; mi < 2; ++mi)
#pragma unroll
    for (int ni = 0; ni < 4; ++ni) c[mi][ni] = (v8f)ZERO8F;

#pragma unroll 2
  for (int k0 = 0; k0 < 1024; k0 += 32) {
    v16bf a0 = load_a16x32(ap + k0, 1024, lane);
    v16bf a1 = load_a16x32(ap + 16 * 1024 + k0, 1024, lane);
    v16bf b0 = load_b32x16(bp + k0, 1024, lane);
    v16bf b1 = load_b32x16(bp + 16 * 1024 + k0, 1024, lane);
    v16bf b2 = load_b32x16(bp + 32 * 1024 + k0, 1024, lane);
    v16bf b3 = load_b32x16(bp + 48 * 1024 + k0, 1024, lane);
    c[0][0] = wmma_bf16(a0, b0, c[0][0]);
    c[0][1] = wmma_bf16(a0, b1, c[0][1]);
    c[0][2] = wmma_bf16(a0, b2, c[0][2]);
    c[0][3] = wmma_bf16(a0, b3, c[0][3]);
    c[1][0] = wmma_bf16(a1, b0, c[1][0]);
    c[1][1] = wmma_bf16(a1, b1, c[1][1]);
    c[1][2] = wmma_bf16(a1, b2, c[1][2]);
    c[1][3] = wmma_bf16(a1, b3, c[1][3]);
  }

  const int half = lane >> 4;
#pragma unroll
  for (int mi = 0; mi < 2; ++mi) {
#pragma unroll
    for (int ni = 0; ni < 4; ++ni) {
      const int n = n0 + ni * 16 + (lane & 15);
      const int h = n >> 7, d = n & 127;
#pragma unroll
      for (int r = 0; r < 8; ++r) {
        int m = m0 + mi * 16 + r + 8 * half;
        int bb = m / 2000, t = m - bb * 2000;
        size_t idx = transposed
            ? ((size_t)(bb * 8 + h) * 128 + d) * 2000 + t
            : ((size_t)(bb * 8 + h) * 2000 + t) * 128 + d;
        dst[idx] = (__bf16)c[mi][ni][r];
      }
    }
  }
}

// ---------------------------------------------------------------------------
// Chunked time-mixing GEMM:
//   out[m,d] = sum_n Wb[c][m,n] * src_t[(bh*128+d)*2000 + c*1000 + n] + bias[c][m]
//   M = 1000 (32 tiles of 32 rows, guarded), N = 128 (2 tiles of 64),
//   K = 1000 (guarded tail).
//   transposed==0: dst (bh, t, d) bf16   (K for attention scores)
//   transposed==1: dst (bh, d, t) bf16   (V for attention output)
// grid: x = 64 tiles / 4 waves, y = bh(32), z = c(2)
// ---------------------------------------------------------------------------
__global__ __launch_bounds__(128) void chunk_gemm(const __bf16* __restrict__ Wb,
                                                  const float* __restrict__ bias,
                                                  const __bf16* __restrict__ src_t,
                                                  __bf16* __restrict__ dst,
                                                  int transposed) {
  const int lane = threadIdx.x & 31;
  const int wave = threadIdx.x >> 5;
  const int flat = blockIdx.x * (blockDim.x >> 5) + wave;
  if (flat >= 32 * 2) return;
  const int mt = flat >> 1, nt = flat & 1;
  const int bh = blockIdx.y, cch = blockIdx.z;
  const int m0 = mt * 32, d0 = nt * 64;

  const __bf16* A = Wb + (size_t)cch * 1000 * 1000;  // 1000 x 1000
  const __bf16* Bcol = src_t + ((size_t)bh * 128 + d0) * 2000 + cch * 1000;

  v8f c[2][4];
#pragma unroll
  for (int mi = 0; mi < 2; ++mi)
#pragma unroll
    for (int ni = 0; ni < 4; ++ni) c[mi][ni] = (v8f)ZERO8F;

  for (int k0 = 0; k0 < 1024; k0 += 32) {
    const bool kfull = (k0 + 32 <= 1000);
    v16bf a[2], b[4];
#pragma unroll
    for (int mi = 0; mi < 2; ++mi) {
      int row0 = m0 + mi * 16;
      a[mi] = (kfull && row0 + 16 <= 1000)
          ? load_a16x32(A + (size_t)row0 * 1000 + k0, 1000, lane)
          : load_a_guard(A, 1000, lane, row0, 1000, k0, 1000);
    }
#pragma unroll
    for (int ni = 0; ni < 4; ++ni) {
      const __bf16* col = Bcol + (size_t)ni * 16 * 2000;
      b[ni] = kfull ? load_b32x16(col + k0, 2000, lane)
                    : load_b_guard(col, 2000, lane, k0, 1000);
    }
#pragma unroll
    for (int mi = 0; mi < 2; ++mi)
#pragma unroll
      for (int ni = 0; ni < 4; ++ni)
        c[mi][ni] = wmma_bf16(a[mi], b[ni], c[mi][ni]);
  }

  const int half = lane >> 4;
  const float* bi = bias + cch * 1000;
#pragma unroll
  for (int mi = 0; mi < 2; ++mi) {
#pragma unroll
    for (int ni = 0; ni < 4; ++ni) {
      const int d = d0 + ni * 16 + (lane & 15);
#pragma unroll
      for (int r = 0; r < 8; ++r) {
        int m = m0 + mi * 16 + r + 8 * half;
        if (m >= 1000) continue;
        float val = c[mi][ni][r] + bi[m];
        size_t t = (size_t)cch * 1000 + m;
        size_t idx = transposed
            ? ((size_t)bh * 128 + d) * 2000 + t
            : ((size_t)bh * 2000 + t) * 128 + d;
        dst[idx] = (__bf16)val;
      }
    }
  }
}

// ---------------------------------------------------------------------------
// Flash attention (non-causal): one wave per 16-query tile.
//   Q: (bh, t, d) bf16, K: (bh, t, d) bf16, Vt: (bh, d, t) bf16
//   out: (b, t, h, d) f32 == (B,T,C)
// j-steps of 32 (62 full + 16-wide tail masked); online softmax; P tile goes
// through LDS to convert C-layout -> A-fragment layout.
// ---------------------------------------------------------------------------
__global__ __launch_bounds__(128) void attn_kernel(const __bf16* __restrict__ Q,
                                                   const __bf16* __restrict__ K,
                                                   const __bf16* __restrict__ Vt,
                                                   float* __restrict__ out) {
  __shared__ __bf16 pbuf[4][16 * 32];
  const int lane = threadIdx.x & 31;
  const int wave = threadIdx.x >> 5;
  const int id = blockIdx.x * 4 + wave;
  if (id >= 32 * 125) return;
  const int bh = id / 125, qt = id - 125 * bh;
  const int i0 = qt * 16;
  const int n = lane & 15, half = lane >> 4;
  const float scale = 0.08838834764831845f;  // 1/sqrt(128)

  // Q A-fragments for the whole 16x128 tile (reused across all key tiles)
  const __bf16* qbase = Q + ((size_t)bh * 2000 + i0) * 128;
  v16bf qf[4];
#pragma unroll
  for (int kk = 0; kk < 4; ++kk) qf[kk] = load_a16x32(qbase + 32 * kk, 128, lane);

  v8f acc[8];
#pragma unroll
  for (int dt = 0; dt < 8; ++dt) acc[dt] = (v8f)ZERO8F;
  float mrow[8], lrow[8];
#pragma unroll
  for (int r = 0; r < 8; ++r) { mrow[r] = -3.0e38f; lrow[r] = 0.f; }

  for (int jt = 0; jt < 63; ++jt) {
    const int j0 = jt * 32;
    const bool full = (j0 + 32 <= 2000);  // tail tile: only first 16 keys valid

    // scores: two 16x16 tiles over d-reduction (4 x K=32)
    v8f s0 = ZERO8F, s1 = ZERO8F;
    const __bf16* kb0 = K + ((size_t)bh * 2000 + j0) * 128;
#pragma unroll
    for (int kk = 0; kk < 4; ++kk)
      s0 = wmma_bf16(qf[kk], load_b32x16(kb0 + 32 * kk, 128, lane), s0);
    if (full) {
      const __bf16* kb1 = kb0 + 16 * 128;
#pragma unroll
      for (int kk = 0; kk < 4; ++kk)
        s1 = wmma_bf16(qf[kk], load_b32x16(kb1 + 32 * kk, 128, lane), s1);
    }

    // online softmax (row stats live per lane-half group of 16 lanes)
    float p0[8], p1[8];
#pragma unroll
    for (int r = 0; r < 8; ++r) {
      float t0 = s0[r] * scale;
      float t1 = full ? s1[r] * scale : -3.0e38f;
      float mx = fmaxf(t0, t1);
#pragma unroll
      for (int off = 1; off < 16; off <<= 1)
        mx = fmaxf(mx, __shfl_xor(mx, off, 32));
      float mnew = fmaxf(mrow[r], mx);
      float corr = __expf(mrow[r] - mnew);
      float e0 = __expf(t0 - mnew);
      float e1 = full ? __expf(t1 - mnew) : 0.f;
      float rs = e0 + e1;
#pragma unroll
      for (int off = 1; off < 16; off <<= 1)
        rs += __shfl_xor(rs, off, 32);
      lrow[r] = lrow[r] * corr + rs;
      mrow[r] = mnew;
      p0[r] = e0;
      p1[r] = e1;
#pragma unroll
      for (int dt = 0; dt < 8; ++dt) acc[dt][r] *= corr;
    }

    // P tile (16x32) C-layout -> LDS -> A-fragment layout (wave-internal, DS in-order)
    __bf16* pw = pbuf[wave];
#pragma unroll
    for (int r = 0; r < 8; ++r) {
      int row = r + 8 * half;
      pw[row * 32 + n] = (__bf16)p0[r];
      pw[row * 32 + 16 + n] = (__bf16)p1[r];
    }
    v16bf pa = load_a16x32(pw, 32, lane);

    // acc += P(16x32) @ V(32x128); V columns (fixed d) contiguous in t
    const __bf16* vb = Vt + (size_t)bh * 128 * 2000 + j0;
#pragma unroll
    for (int dt = 0; dt < 8; ++dt) {
      v16bf bfrag = load_b32x16(vb + (size_t)dt * 16 * 2000, 2000, lane);
      acc[dt] = wmma_bf16(pa, bfrag, acc[dt]);
    }
  }

  // epilogue: y = acc / l, scatter to (B, T, nh, hs)
  const int b = bh >> 3, h = bh & 7;
#pragma unroll
  for (int dt = 0; dt < 8; ++dt) {
#pragma unroll
    for (int r = 0; r < 8; ++r) {
      int i = i0 + r + 8 * half;
      float y = acc[dt][r] / lrow[r];
      out[(((size_t)b * 2000 + i) * 8 + h) * 128 + dt * 16 + n] = y;
    }
  }
}

// ---------------------------------------------------------------------------
// Host launch
// ---------------------------------------------------------------------------
extern "C" void kernel_launch(void* const* d_in, const int* in_sizes, int n_in,
                              void* d_out, int out_size, void* d_ws, size_t ws_size,
                              hipStream_t stream) {
  const float* x    = (const float*)d_in[0];  // (4,2000,1024)
  const float* Wq   = (const float*)d_in[1];  // (1024,1024)
  const float* Wk   = (const float*)d_in[2];
  const float* Wv   = (const float*)d_in[3];
  const float* pk_W = (const float*)d_in[4];  // (2,1000,1000)
  const float* pk_b = (const float*)d_in[5];  // (2,1000)
  const float* pv_W = (const float*)d_in[6];
  const float* pv_b = (const float*)d_in[7];
  float* out = (float*)d_out;                 // (4,2000,1024)

  const size_t NX = 8192000;   // 8000*1024
  const size_t NW = 1048576;   // 1024*1024
  const size_t NP = 2000000;   // 2*1000*1000

  char* ws = (char*)d_ws;
  size_t off = 0;
  auto alloc = [&](size_t bytes) {
    size_t cur = off;
    off += (bytes + 255) & ~(size_t)255;
    return cur;
  };
  __bf16* xb   = (__bf16*)(ws + alloc(NX * 2));
  __bf16* wqb  = (__bf16*)(ws + alloc(NW * 2));
  __bf16* wkb  = (__bf16*)(ws + alloc(NW * 2));
  __bf16* wvb  = (__bf16*)(ws + alloc(NW * 2));
  __bf16* pkwb = (__bf16*)(ws + alloc(NP * 2));
  __bf16* pvwb = (__bf16*)(ws + alloc(NP * 2));
  __bf16* qb   = (__bf16*)(ws + alloc(NX * 2));
  __bf16* tmpb = (__bf16*)(ws + alloc(NX * 2));        // projected K/V in (bh,d,t)
  __bf16* kb   = (__bf16*)(ws + alloc(NX * 2));        // mixed K  (bh,t,d)
  __bf16* vtb  = (__bf16*)(ws + alloc(NX * 2 + 256));  // mixed V (bh,d,t) + tail pad

  // f32 -> bf16 conversions
  cvt_f32_bf16<<<(int)((NX + 255) / 256), 256, 0, stream>>>(x, xb, (int)NX);
  cvt_f32_bf16<<<(int)((NW + 255) / 256), 256, 0, stream>>>(Wq, wqb, (int)NW);
  cvt_f32_bf16<<<(int)((NW + 255) / 256), 256, 0, stream>>>(Wk, wkb, (int)NW);
  cvt_f32_bf16<<<(int)((NW + 255) / 256), 256, 0, stream>>>(Wv, wvb, (int)NW);
  cvt_f32_bf16<<<(int)((NP + 255) / 256), 256, 0, stream>>>(pk_W, pkwb, (int)NP);
  cvt_f32_bf16<<<(int)((NP + 255) / 256), 256, 0, stream>>>(pv_W, pvwb, (int)NP);

  // projections (250*16 tiles of 32x64, 8 waves/block)
  proj_gemm<<<500, 256, 0, stream>>>(xb, wqb, qb, 0);
  proj_gemm<<<500, 256, 0, stream>>>(xb, wkb, tmpb, 1);
  chunk_gemm<<<dim3(16, 32, 2), 128, 0, stream>>>(pkwb, pk_b, tmpb, kb, 0);
  proj_gemm<<<500, 256, 0, stream>>>(xb, wvb, tmpb, 1);
  chunk_gemm<<<dim3(16, 32, 2), 128, 0, stream>>>(pvwb, pv_b, tmpb, vtb, 1);

  // attention: 4000 query tiles, 4 waves/block
  attn_kernel<<<1000, 128, 0, stream>>>(qb, kb, vtb, out);
}
